// SuperGAT_L2_intervention_66271345377526
// MI455X (gfx1250) — matrix operand order repaired and checked by reading, not compiled, not run on hardware
//
#include <hip/hip_runtime.h>

// ---------------------------------------------------------------------------
// SuperGAT 2-layer forward for MI455X (gfx1250, wave32, WMMA).
// GEMMs run on v_wmma_f32_16x16x32_bf16 (fp32 accumulate), one wave per
// 16x(16*TN) output strip so each A fragment feeds TN WMMAs. Edge phases are
// bandwidth-bound gather/scatter with f32 atomics.
// ---------------------------------------------------------------------------

#define NN    50000
#define EE    400000
#define ETOT  (EE + NN)      // 450000 after self-loops
#define INC   512
#define HID   64
#define OUTC  40
#define HH    4
#define HC1   (HH * HID)     // 256
#define HC2   (HH * OUTC)    // 160
#define NEG_SLOPE 0.2f
#define EPSD  1e-16f

typedef __attribute__((ext_vector_type(16))) __bf16 v16bf;
typedef __attribute__((ext_vector_type(8)))  float  v8f;

union BFrag { unsigned int u[8]; v16bf v; };

// ------------------------------- helpers -----------------------------------

__device__ __forceinline__ unsigned short f2bf(float f) {
  unsigned int u = __float_as_uint(f);
  unsigned int r = 0x7FFFu + ((u >> 16) & 1u);   // round to nearest even
  return (unsigned short)((u + r) >> 16);
}

__device__ __forceinline__ void atomicMaxFloat(float* addr, float val) {
  // int/uint punned max: works for all sign combinations (init = -inf).
  if (val >= 0.f) atomicMax((int*)addr, __float_as_int(val));
  else            atomicMin((unsigned int*)addr, __float_as_uint(val));
}

// --------------------------- conversion kernels ----------------------------

__global__ void k_fill_f32(float* __restrict__ p, float v, long long n) {
  long long t = (long long)blockIdx.x * blockDim.x + threadIdx.x;
  if (t < n) p[t] = v;
}

__global__ void k_f32_to_bf16(const float* __restrict__ src,
                              unsigned short* __restrict__ dst, long long n) {
  long long t = (long long)blockIdx.x * blockDim.x + threadIdx.x;
  if (t < n) dst[t] = f2bf(src[t]);
}

// Wt[n*K + k] = bf16(W[k*N + n])   (transpose so B fragments load contiguous K)
__global__ void k_transpose_bf16(const float* __restrict__ W,
                                 unsigned short* __restrict__ Wt, int K, int N) {
  int t = blockIdx.x * blockDim.x + threadIdx.x;
  if (t >= K * N) return;
  int n = t / K, k = t - n * K;
  Wt[t] = f2bf(W[(size_t)k * N + n]);
}

// ------------------------------- WMMA GEMM ---------------------------------
// C[M,N] = A[M,K] (bf16 row-major) * Bt[N,K]^T (bf16, pre-transposed weights)
// One wave owns a 16 x (16*TN) output strip: the A fragment is loaded once
// per K-step and reused by TN WMMAs (A L2 re-reads cut by TN, WMMA density up).
// Fragment layouts follow CDNA5 ISA 7.12.2 (wave32).

template <int TN>
__global__ void k_wmma_gemm_bf16(const unsigned short* __restrict__ A,
                                 const unsigned short* __restrict__ Bt,
                                 float* __restrict__ C,
                                 int M, int N, int K) {
  int gw   = (int)((blockIdx.x * blockDim.x + threadIdx.x) >> 5);
  int lane = threadIdx.x & 31;
  int mt   = M >> 4;
  int ngrp = N / (16 * TN);
  if (gw >= mt * ngrp) return;               // uniform per wave: EXEC stays full
  int tm = gw / ngrp, tg = gw - tm * ngrp;
  int hi = lane >> 4;                        // 0: lanes 0-15, 1: lanes 16-31
  int lr = lane & 15;

  const unsigned short* arow  = A  + (size_t)(tm * 16 + lr) * K;
  const unsigned short* brow0 = Bt + (size_t)(tg * TN * 16 + lr) * K;

  v8f acc[TN];
#pragma unroll
  for (int t = 0; t < TN; ++t) acc[t] = (v8f){};

  for (int k0 = 0; k0 < K; k0 += 32) {
    BFrag a;
#pragma unroll
    for (int r = 0; r < 8; ++r) {
      // A 16x32 bf16: VGPR0-3 -> K = 2r + 8*hi ; VGPR4-7 -> K = 16 + 2(r-4) + 8*hi
      int ka = k0 + ((r < 4) ? (2 * r + 8 * hi) : (8 + 2 * r + 8 * hi));
      a.u[r] = *(const unsigned int*)(arow + ka);
    }
#pragma unroll
    for (int t = 0; t < TN; ++t) {
      BFrag b;
#pragma unroll
      for (int r = 0; r < 8; ++r) {
        // B 32x16 bf16: same K order as A: VGPR0-3 -> K = 2r + 8*hi ; VGPR4-7 -> K = 16 + 2(r-4) + 8*hi
        int kb = k0 + ((r < 4) ? (2 * r + 8 * hi) : (8 + 2 * r + 8 * hi));
        b.u[r] = *(const unsigned int*)(brow0 + (size_t)t * 16 * K + kb);
      }
      acc[t] = __builtin_amdgcn_wmma_f32_16x16x32_bf16(
          false, a.v, false, b.v, (short)0, acc[t], false, false);
    }
  }

  // D 16x16 f32: VGPR r -> row (r + 8*hi), col = lane&15
#pragma unroll
  for (int t = 0; t < TN; ++t) {
    float* cbase = C + (size_t)(tm * 16) * N + (tg * TN + t) * 16 + lr;
#pragma unroll
    for (int r = 0; r < 8; ++r) cbase[(size_t)(r + 8 * hi) * N] = acc[t][r];
  }
}

// ---------------------------- edge-phase kernels ---------------------------
// One wave per (edge, head): dot over C channels, shuffle-reduce, leaky-relu,
// atomic segment-max into amax[dst,h]. Self-loops are edges e >= EE.

__global__ void k_edge_logits(const float* __restrict__ Hf,     // [N, H*C]
                              const long long* __restrict__ srcA,
                              const long long* __restrict__ dstA,
                              float* __restrict__ alphaBuf,     // [ETOT, H]
                              float* __restrict__ amax,         // [N, H]
                              int C) {
  int gw   = (int)((blockIdx.x * blockDim.x + threadIdx.x) >> 5);
  int lane = threadIdx.x & 31;
  if (gw >= ETOT * HH) return;
  int e = gw / HH, h = gw - e * HH;
  long long s, d;
  if (e < EE) { s = srcA[e]; d = dstA[e]; } else { s = d = (long long)(e - EE); }
  const float* pi = Hf + (size_t)d * (HH * C) + (size_t)h * C;
  const float* pj = Hf + (size_t)s * (HH * C) + (size_t)h * C;
  float sum = 0.f;
  for (int c = lane; c < C; c += 32) sum += pi[c] * pj[c];
#pragma unroll
  for (int off = 16; off > 0; off >>= 1) sum += __shfl_xor(sum, off, 32);
  if (lane == 0) {
    float a = sum > 0.f ? sum : NEG_SLOPE * sum;
    alphaBuf[(size_t)e * HH + h] = a;
    atomicMaxFloat(&amax[(size_t)d * HH + h], a);
  }
}

// One thread per (edge, head): exp(alpha - amax[dst]), segment-sum denom.
__global__ void k_edge_expsum(const long long* __restrict__ dstA,
                              float* __restrict__ alphaBuf,
                              const float* __restrict__ amax,
                              float* __restrict__ denom) {
  int t = blockIdx.x * blockDim.x + threadIdx.x;
  if (t >= ETOT * HH) return;
  int e = t / HH, h = t - e * HH;
  long long d = (e < EE) ? dstA[e] : (long long)(e - EE);
  float a = __expf(alphaBuf[t] - amax[(size_t)d * HH + h]);
  alphaBuf[t] = a;
  atomicAdd(&denom[(size_t)d * HH + h], a);
}

// One thread per (edge, channel j of H*C): agg[dst, j] += Hf[src, j] * w.
__global__ void k_edge_aggregate(const float* __restrict__ Hf,
                                 const long long* __restrict__ srcA,
                                 const long long* __restrict__ dstA,
                                 const float* __restrict__ alphaBuf,
                                 const float* __restrict__ denom,
                                 float* __restrict__ agg, int C) {
  long long t = (long long)blockIdx.x * blockDim.x + threadIdx.x;
  int HC = HH * C;
  if (t >= (long long)ETOT * HC) return;
  int e = (int)(t / HC), j = (int)(t - (long long)e * HC);
  int h = j / C;
  long long s, d;
  if (e < EE) { s = srcA[e]; d = dstA[e]; } else { s = d = (long long)(e - EE); }
  float w = alphaBuf[(size_t)e * HH + h] / (denom[(size_t)d * HH + h] + EPSD);
  atomicAdd(&agg[(size_t)d * HC + j], Hf[(size_t)s * HC + j] * w);
}

// Layer-1 epilogue: +bias, ELU, convert to bf16 for GEMM2.
__global__ void k_epilogue1(const float* __restrict__ agg,
                            const float* __restrict__ b1,
                            unsigned short* __restrict__ Hact, long long total) {
  long long t = (long long)blockIdx.x * blockDim.x + threadIdx.x;
  if (t >= total) return;
  float v = agg[t] + b1[(int)(t % HC1)];
  v = v > 0.f ? v : (__expf(v) - 1.f);       // ELU, alpha=1
  Hact[t] = f2bf(v);
}

// Layer-2 epilogue: mean over heads, +bias, log_softmax -> d_out.
__global__ void k_epilogue2(const float* __restrict__ agg2,
                            const float* __restrict__ b2,
                            float* __restrict__ out) {
  int n = blockIdx.x * blockDim.x + threadIdx.x;
  if (n >= NN) return;
  float v[OUTC];
#pragma unroll
  for (int c = 0; c < OUTC; ++c) {
    float s = 0.f;
#pragma unroll
    for (int h = 0; h < HH; ++h) s += agg2[(size_t)n * HC2 + h * OUTC + c];
    v[c] = s * (1.f / HH) + b2[c];
  }
  float mx = v[0];
#pragma unroll
  for (int c = 1; c < OUTC; ++c) mx = fmaxf(mx, v[c]);
  float se = 0.f;
#pragma unroll
  for (int c = 0; c < OUTC; ++c) se += __expf(v[c] - mx);
  float lse = mx + __logf(se);
#pragma unroll
  for (int c = 0; c < OUTC; ++c) out[(size_t)n * OUTC + c] = v[c] - lse;
}

__global__ void k_att_loss(float* __restrict__ out) {
  if (blockIdx.x == 0 && threadIdx.x == 0) out[(size_t)NN * OUTC] = 0.f;
}

// ------------------------------- launcher ----------------------------------

static inline unsigned int cdivu(long long a, long long b) {
  return (unsigned int)((a + b - 1) / b);
}

extern "C" void kernel_launch(void* const* d_in, const int* in_sizes, int n_in,
                              void* d_out, int out_size, void* d_ws, size_t ws_size,
                              hipStream_t stream) {
  (void)in_sizes; (void)n_in; (void)out_size; (void)ws_size;

  const float*     x   = (const float*)d_in[0];
  const long long* ei  = (const long long*)d_in[1];   // int64 [2, E]
  const float*     W1  = (const float*)d_in[2];       // [512, 256]
  const float*     b1  = (const float*)d_in[3];       // [256]
  const float*     W2  = (const float*)d_in[4];       // [256, 160]
  const float*     b2  = (const float*)d_in[5];       // [40]
  float*           out = (float*)d_out;

  const long long* srcA = ei;
  const long long* dstA = ei + EE;

  // ---- workspace layout (bytes, 256-aligned; regions reused across phases)
  char* ws = (char*)d_ws;
  const size_t OFF_XBF   = 0;                              // 51,200,000  X bf16 [N,512]; reused for AGG1
  const size_t OFF_W1T   = 51200000;                       //    262,144  W1^T bf16 [256,512]
  const size_t OFF_H1    = 51462144;                       // 51,200,000  H1 f32 [N,256]; reused for H2 f32 [N,160]
  const size_t OFF_AMAX  = 102662144;                      //    800,000  amax f32 [N,4] (both layers)
  const size_t OFF_DENOM = 103462144;                      //    800,000  denom f32 [N,4]
  const size_t OFF_ALPHA = 104262144;                      //  7,200,000  alpha f32 [ETOT,4]
  const size_t OFF_H1ACT = 111462144;                      // 25,600,000  ELU(H1) bf16 [N,256]
  const size_t OFF_W2T   = 137062144;                      //     81,920  W2^T bf16 [160,256]
  const size_t OFF_AGG2  = 137144064;                      // 32,000,000  agg2 f32 [N,160]  (total ~169 MB)

  unsigned short* XBF   = (unsigned short*)(ws + OFF_XBF);
  unsigned short* W1T   = (unsigned short*)(ws + OFF_W1T);
  float*          H1    = (float*)(ws + OFF_H1);
  float*          AMAX  = (float*)(ws + OFF_AMAX);
  float*          DENOM = (float*)(ws + OFF_DENOM);
  float*          ALPHA = (float*)(ws + OFF_ALPHA);
  float*          AGG1  = (float*)(ws + OFF_XBF);          // reuse XBF after GEMM1
  unsigned short* H1ACT = (unsigned short*)(ws + OFF_H1ACT);
  unsigned short* W2T   = (unsigned short*)(ws + OFF_W2T);
  float*          H2    = (float*)(ws + OFF_H1);           // reuse H1 region
  float*          AGG2  = (float*)(ws + OFF_AGG2);

  const int BLK = 256;
  const float NEG_INF = __builtin_huge_valf() * -1.f;

  // ---------------- layer 1 ----------------
  // convert X and W1/W2 to bf16 (weights transposed to [N,K])
  k_f32_to_bf16 <<<cdivu((long long)NN * INC, BLK), BLK, 0, stream>>>(x, XBF, (long long)NN * INC);
  k_transpose_bf16<<<cdivu((long long)INC * HC1, BLK), BLK, 0, stream>>>(W1, W1T, INC, HC1);
  k_transpose_bf16<<<cdivu((long long)HC1 * HC2, BLK), BLK, 0, stream>>>(W2, W2T, HC1, HC2);

  // GEMM1: [50000,512] x [512,256] -> H1 ; TN=4 => 16x64 strip, 3125*4 waves
  {
    long long waves = (long long)(NN / 16) * (HC1 / 64);
    k_wmma_gemm_bf16<4><<<cdivu(waves * 32, BLK), BLK, 0, stream>>>(XBF, W1T, H1, NN, HC1, INC);
  }

  // segment-softmax state init (AGG1 overwrites the XBF region — X is dead)
  k_fill_f32<<<cdivu((long long)NN * HH, BLK), BLK, 0, stream>>>(AMAX, NEG_INF, (long long)NN * HH);
  k_fill_f32<<<cdivu((long long)NN * HH, BLK), BLK, 0, stream>>>(DENOM, 0.f, (long long)NN * HH);
  k_fill_f32<<<cdivu((long long)NN * HC1, BLK), BLK, 0, stream>>>(AGG1, 0.f, (long long)NN * HC1);

  k_edge_logits   <<<cdivu((long long)ETOT * HH * 32, BLK), BLK, 0, stream>>>(H1, srcA, dstA, ALPHA, AMAX, HID);
  k_edge_expsum   <<<cdivu((long long)ETOT * HH, BLK), BLK, 0, stream>>>(dstA, ALPHA, AMAX, DENOM);
  k_edge_aggregate<<<cdivu((long long)ETOT * HC1, BLK), BLK, 0, stream>>>(H1, srcA, dstA, ALPHA, DENOM, AGG1, HID);

  k_epilogue1<<<cdivu((long long)NN * HC1, BLK), BLK, 0, stream>>>(AGG1, b1, H1ACT, (long long)NN * HC1);

  // ---------------- layer 2 ----------------
  // GEMM2: [50000,256] x [256,160] -> H2 ; TN=5 => 16x80 strip, 3125*2 waves
  {
    long long waves = (long long)(NN / 16) * (HC2 / 80);
    k_wmma_gemm_bf16<5><<<cdivu(waves * 32, BLK), BLK, 0, stream>>>(H1ACT, W2T, H2, NN, HC2, HC1);
  }

  k_fill_f32<<<cdivu((long long)NN * HH, BLK), BLK, 0, stream>>>(AMAX, NEG_INF, (long long)NN * HH);
  k_fill_f32<<<cdivu((long long)NN * HH, BLK), BLK, 0, stream>>>(DENOM, 0.f, (long long)NN * HH);
  k_fill_f32<<<cdivu((long long)NN * HC2, BLK), BLK, 0, stream>>>(AGG2, 0.f, (long long)NN * HC2);

  k_edge_logits   <<<cdivu((long long)ETOT * HH * 32, BLK), BLK, 0, stream>>>(H2, srcA, dstA, ALPHA, AMAX, OUTC);
  k_edge_expsum   <<<cdivu((long long)ETOT * HH, BLK), BLK, 0, stream>>>(dstA, ALPHA, AMAX, DENOM);
  k_edge_aggregate<<<cdivu((long long)ETOT * HC2, BLK), BLK, 0, stream>>>(H2, srcA, dstA, ALPHA, DENOM, AGG2, OUTC);

  k_epilogue2<<<cdivu(NN, BLK), BLK, 0, stream>>>(AGG2, b2, out);
  k_att_loss <<<1, 1, 0, stream>>>(out);
}
